// DecoderRNNT_74947179315293
// MI455X (gfx1250) — matrix operand order repaired and checked by reading, not compiled, not run on hardware
//
#include <hip/hip_runtime.h>
#include <hip/hip_fp16.h>

#define BLANK  0
#define B_     4
#define T_     128
#define U_     64
#define EPROJS 512
#define EMBED_ 512
#define DUNITS 1024
#define JOINT_ 512
#define ODIM_  2048
#define GATES  (4*DUNITS)   // 4096

typedef __attribute__((ext_vector_type(16))) _Float16 hvec16;
typedef __attribute__((ext_vector_type(8)))  _Float16 hvec8;
typedef __attribute__((ext_vector_type(8)))  float    fvec8;

// ---------------------------------------------------------------- helpers ---

__device__ __forceinline__ fvec8 zero8() { fvec8 z = {}; return z; }

__device__ __forceinline__ fvec8 wmma16(hvec16 a, hvec16 b, fvec8 c) {
  // D = A(16x32 f16) x B(32x16 f16) + C(16x16 f32)
  return __builtin_amdgcn_wmma_f32_16x16x32_f16(
      /*neg_a=*/false, a, /*neg_b=*/false, b,
      /*c_mod=*/(short)0, c, /*reuse_a=*/false, /*reuse_b=*/false);
}

// A-fragment: row-major A (lda elems), tile origin (m0,k0); rows clamped to mMax-1
// lane layout: M = lane&15 ; element e -> K = (e<8?0:16) + (lane<16?0:8) + (e&7)
__device__ __forceinline__ hvec16 load_a(const _Float16* A, int lda, int m0,
                                         int k0, int mMax, int lane) {
  int m = m0 + (lane & 15);
  if (m >= mMax) m = mMax - 1;        // duplicate row; pollutes only unused C rows
  const _Float16* p = A + (size_t)m * lda + k0 + ((lane >> 4) << 3);
  hvec8 lo = *(const hvec8*)(p);
  hvec8 hi = *(const hvec8*)(p + 16);
  hvec16 r;
#pragma unroll
  for (int i = 0; i < 8; ++i) { r[i] = lo[i]; r[i + 8] = hi[i]; }
  return r;
}

// B-fragment for C = A * W^T with W row-major (N x K): B(k,n) = W[n,k]
// lane layout: N = lane&15 ; element e -> K = (e<8?0:16) + (lane<16?0:8) + (e&7)
__device__ __forceinline__ hvec16 load_b(const _Float16* W, int ldk, int n0,
                                         int k0, int lane) {
  const _Float16* p = W + (size_t)(n0 + (lane & 15)) * ldk + k0 + ((lane >> 4) << 3);
  hvec8 lo = *(const hvec8*)(p);
  hvec8 hi = *(const hvec8*)(p + 16);
  hvec16 r;
#pragma unroll
  for (int i = 0; i < 8; ++i) { r[i] = lo[i]; r[i + 8] = hi[i]; }
  return r;
}

// fast tanh: 1 hardware exp + fast divide; clamp avoids inf/inf.
// |error| ~1e-6 relative -- far below the f16 quantization applied after.
__device__ __forceinline__ float tanh_fast(float x) {
  float cx = fminf(fmaxf(x, -15.f), 15.f);
  float e = __expf(2.f * cx);
  return __fdividef(e - 1.f, e + 1.f);
}

__device__ __forceinline__ float sigm(float x) {
  return __fdividef(1.0f, 1.0f + __expf(-x));
}

// grid-wide spin barrier (all blocks resident; counter monotonically increases)
__device__ __forceinline__ void grid_barrier(int* bar, int goal) {
  __syncthreads();
  __threadfence();
  if (threadIdx.x == 0) {
    atomicAdd(bar, 1);
    while (__hip_atomic_load(bar, __ATOMIC_ACQUIRE, __HIP_MEMORY_SCOPE_AGENT) < goal) {
      __builtin_amdgcn_s_sleep(1);
    }
  }
  __syncthreads();
}

// ------------------------------------------------------------- tiny kernels ---

__global__ void zero_i32_kernel(int* p, int n) {
  int i = blockIdx.x * blockDim.x + threadIdx.x;
  if (i < n) p[i] = 0;
}

__global__ void zero_f16_kernel(_Float16* p, int n) {
  for (int i = blockIdx.x * blockDim.x + threadIdx.x; i < n;
       i += gridDim.x * blockDim.x)
    p[i] = (_Float16)0.0f;
}

__global__ void cvt_f16_kernel(const float* __restrict__ s,
                               _Float16* __restrict__ d, int n) {
  for (int i = blockIdx.x * blockDim.x + threadIdx.x; i < n;
       i += gridDim.x * blockDim.x)
    d[i] = (_Float16)s[i];
}

__global__ void add_f32_kernel(const float* __restrict__ a,
                               const float* __restrict__ b,
                               float* __restrict__ c, int n) {
  int i = blockIdx.x * blockDim.x + threadIdx.x;
  if (i < n) c[i] = a[i] + b[i];
}

// eys = embed[ys] * (ys != BLANK) -> f16, (B*U, EMBED)
__global__ void embed_kernel(const int* __restrict__ ys,
                             const float* __restrict__ embed,
                             _Float16* __restrict__ eys, int total) {
  for (int idx = blockIdx.x * blockDim.x + threadIdx.x; idx < total;
       idx += gridDim.x * blockDim.x) {
    int e = idx & (EMBED_ - 1);
    int bu = idx >> 9;                 // EMBED_ == 512
    int id = ys[bu];
    float v = (id == BLANK) ? 0.0f : embed[(size_t)id * EMBED_ + e];
    eys[idx] = (_Float16)v;
  }
}

// ------------------------------------------------- generic C = A*W^T + bias ---
// A: (M,K) f16 row-major; W: (N,K) f16 row-major; C: (M,N) f32.
// One 16x16 C tile per wave, grid-stride over tiles.
__global__ __launch_bounds__(256)
void gemm_awt_kernel(const _Float16* __restrict__ A,
                     const _Float16* __restrict__ W,
                     const float* __restrict__ bias,
                     float* __restrict__ C, int M, int N, int K) {
  const int lane = threadIdx.x & 31;
  const int gw = blockIdx.x * (blockDim.x >> 5) + (threadIdx.x >> 5);
  const int nw = gridDim.x * (blockDim.x >> 5);
  const int mt_n = (M + 15) >> 4;
  const int nt_n = N >> 4;
  for (int t = gw; t < mt_n * nt_n; t += nw) {
    const int m0 = (t / nt_n) << 4;
    const int n0 = (t % nt_n) << 4;
    fvec8 acc = zero8();
    for (int kb = 0; kb < K; kb += 32) {
      hvec16 a = load_a(A, K, m0, kb, M, lane);
      hvec16 b = load_b(W, K, n0, kb, lane);
      acc = wmma16(a, b, acc);
    }
    const int n = n0 + (lane & 15);
    const float bv = bias ? bias[n] : 0.0f;
#pragma unroll
    for (int v = 0; v < 8; ++v) {
      const int m = m0 + v + ((lane >> 4) << 3);
      if (m < M) C[(size_t)m * N + n] = acc[v] + bv;
    }
  }
}

// ------------------------------------------------------- persistent LSTM scan ---
// 8 blocks x 256 threads = 64 waves; wave w owns hidden units [16w, 16w+16).
// Cell state c kept in registers for the whole scan (lanes 0..15 hold b=0..3 in
// accumulator VGPRs 0..3). h ping-pongs through tiny f16 global buffers.
__global__ __launch_bounds__(256)
void lstm_scan_kernel(const float* __restrict__ X0,        // (B*U, 4096) precomputed
                      const _Float16* __restrict__ Whh0,   // (4096,1024)
                      const _Float16* __restrict__ Wih1,   // (4096,1024)
                      const _Float16* __restrict__ Whh1,   // (4096,1024)
                      const float* __restrict__ bias1,     // (4096) b_ih1+b_hh1
                      _Float16* __restrict__ Hbuf,         // 4 bufs x (4*DUNITS)
                      _Float16* __restrict__ hdec,         // (B*U, DUNITS) f16
                      int* bar) {
  const int lane = threadIdx.x & 31;
  const int wgid = blockIdx.x * (blockDim.x >> 5) + (threadIdx.x >> 5); // 0..63
  const int j0 = wgid << 4;
  const int jl = j0 + (lane & 15);
  _Float16* H0[2] = { Hbuf,               Hbuf + 4 * DUNITS };
  _Float16* H1[2] = { Hbuf + 8 * DUNITS,  Hbuf + 12 * DUNITS };
  float c0[4] = {0.f, 0.f, 0.f, 0.f};
  float c1[4] = {0.f, 0.f, 0.f, 0.f};
  int ep = 0;
  const int nb = gridDim.x;

  for (int u = 0; u < U_; ++u) {
    const int p = u & 1;

    // ---- layer 0: gates = X0[:,u,:] + h0_old @ Whh0^T --------------------
    {
      fvec8 acc[4];
#pragma unroll
      for (int g = 0; g < 4; ++g) acc[g] = zero8();
      const _Float16* h0o = H0[p];
#pragma unroll 4
      for (int kb = 0; kb < DUNITS; kb += 32) {
        hvec16 a = load_a(h0o, DUNITS, 0, kb, B_, lane);
#pragma unroll
        for (int g = 0; g < 4; ++g) {
          hvec16 b = load_b(Whh0, DUNITS, g * DUNITS + j0, kb, lane);
          acc[g] = wmma16(a, b, acc[g]);
        }
      }
      if (lane < 16) {
#pragma unroll
        for (int bb = 0; bb < B_; ++bb) {
          const float* x0r = X0 + (size_t)(bb * U_ + u) * GATES;
          float gi = acc[0][bb] + x0r[0 * DUNITS + jl];
          float gf = acc[1][bb] + x0r[1 * DUNITS + jl];
          float gg = acc[2][bb] + x0r[2 * DUNITS + jl];
          float go = acc[3][bb] + x0r[3 * DUNITS + jl];
          float cn = sigm(gf) * c0[bb] + sigm(gi) * tanh_fast(gg);
          c0[bb] = cn;
          H0[p ^ 1][bb * DUNITS + jl] = (_Float16)(sigm(go) * tanh_fast(cn));
        }
      }
    }
    grid_barrier(bar, (++ep) * nb);

    // ---- layer 1: gates = h0_new @ Wih1^T + h1_old @ Whh1^T + bias1 ------
    {
      fvec8 acc[4];
#pragma unroll
      for (int g = 0; g < 4; ++g) acc[g] = zero8();
      const _Float16* h0n = H0[p ^ 1];
      const _Float16* h1o = H1[p];
#pragma unroll 2
      for (int kb = 0; kb < DUNITS; kb += 32) {
        hvec16 a0 = load_a(h0n, DUNITS, 0, kb, B_, lane);
        hvec16 a1 = load_a(h1o, DUNITS, 0, kb, B_, lane);
#pragma unroll
        for (int g = 0; g < 4; ++g) {
          hvec16 b0 = load_b(Wih1, DUNITS, g * DUNITS + j0, kb, lane);
          acc[g] = wmma16(a0, b0, acc[g]);
          hvec16 b1 = load_b(Whh1, DUNITS, g * DUNITS + j0, kb, lane);
          acc[g] = wmma16(a1, b1, acc[g]);
        }
      }
      if (lane < 16) {
#pragma unroll
        for (int bb = 0; bb < B_; ++bb) {
          float gi = acc[0][bb] + bias1[0 * DUNITS + jl];
          float gf = acc[1][bb] + bias1[1 * DUNITS + jl];
          float gg = acc[2][bb] + bias1[2 * DUNITS + jl];
          float go = acc[3][bb] + bias1[3 * DUNITS + jl];
          float cn = sigm(gf) * c1[bb] + sigm(gi) * tanh_fast(gg);
          c1[bb] = cn;
          float hn = sigm(go) * tanh_fast(cn);
          H1[p ^ 1][bb * DUNITS + jl] = (_Float16)hn;
          hdec[(size_t)(bb * U_ + u) * DUNITS + jl] = (_Float16)hn;
        }
      }
    }
    grid_barrier(bar, (++ep) * nb);
  }
}

// ----------------------------------------------------------------- joint ---
// out[b,t,u,:] = tanh(enc_p[b,t,:] + dec_p[b,u,:]) @ Wout^T + b_out
// One WG per (b,t): full 64 u-rows staged as f16 in LDS (64 x 520 = 66.5 KB;
// CDNA5 WGP has 320KB LDS; row stride 520 -> <=2-way bank access).
// A-frags for M-tiles 0..2 preloaded to VGPRs (384 VGPRs, still 2 waves/SIMD),
// only M-tile 3 streamed from LDS (~halves ds bandwidth demand).
// Each W_out B-frag load feeds 4 WMMAs; W_out read once per WG (1GB L2 total).
#define LDS_STRIDE 520
__global__ __launch_bounds__(256)
void joint_kernel(const float* __restrict__ enc_p,   // (B*T, 512)
                  const float* __restrict__ dec_p,   // (B*U, 512)
                  const _Float16* __restrict__ Wout, // (2048, 512)
                  const float* __restrict__ b_out,   // (2048)
                  float* __restrict__ out) {         // (B,T,U,2048)
  __shared__ alignas(16) _Float16 As[64 * LDS_STRIDE];
  const int bt = blockIdx.x;          // 0 .. B*T-1
  const int b = bt / T_;
  const float* er = enc_p + (size_t)bt * JOINT_;
  // stage A = tanh(enc + dec) as f16, 2 elements per iteration
  for (int idx = threadIdx.x; idx < 64 * (JOINT_ / 2); idx += blockDim.x) {
    const int u = idx >> 8;
    const int j2 = (idx & 255) << 1;
    const float2 ev = *(const float2*)(er + j2);
    const float2 dv = *(const float2*)(dec_p + (size_t)(b * U_ + u) * JOINT_ + j2);
    As[u * LDS_STRIDE + j2]     = (_Float16)tanh_fast(ev.x + dv.x);
    As[u * LDS_STRIDE + j2 + 1] = (_Float16)tanh_fast(ev.y + dv.y);
  }
  __syncthreads();

  const int lane = threadIdx.x & 31;
  const int wv = threadIdx.x >> 5;  // 0..7

  // preload A fragments for M-tiles 0..2 (u rows 0..47): 3 x 16 x 8 VGPRs
  hvec16 af[3][16];
#pragma unroll
  for (int mt = 0; mt < 3; ++mt)
#pragma unroll
    for (int kb = 0; kb < 16; ++kb)
      af[mt][kb] = load_a(As, LDS_STRIDE, mt * 16, kb * 32, 64, lane);

  float* orow = out + (size_t)bt * U_ * ODIM_;
  for (int i = 0; i < 16; ++i) {
    const int n0 = ((wv << 4) + i) << 4;
    __builtin_prefetch(Wout + (size_t)n0 * JOINT_, 0, 0);
    fvec8 acc[4];
#pragma unroll
    for (int mt = 0; mt < 4; ++mt) acc[mt] = zero8();
#pragma unroll
    for (int kb = 0; kb < 16; ++kb) {
      hvec16 bf = load_b(Wout, JOINT_, n0, kb * 32, lane);
      acc[0] = wmma16(af[0][kb], bf, acc[0]);
      acc[1] = wmma16(af[1][kb], bf, acc[1]);
      acc[2] = wmma16(af[2][kb], bf, acc[2]);
      hvec16 a3 = load_a(As, LDS_STRIDE, 48, kb * 32, 64, lane);
      acc[3] = wmma16(a3, bf, acc[3]);
    }
    const int o = n0 + (lane & 15);
    const float bo = b_out[o];
#pragma unroll
    for (int mt = 0; mt < 4; ++mt)
#pragma unroll
      for (int v = 0; v < 8; ++v) {
        const int u = (mt << 4) + v + ((lane >> 4) << 3);
        orow[(size_t)u * ODIM_ + o] = acc[mt][v] + bo;
      }
  }
}

// ---------------------------------------------------------------- launcher ---

extern "C" void kernel_launch(void* const* d_in, const int* in_sizes, int n_in,
                              void* d_out, int out_size, void* d_ws,
                              size_t ws_size, hipStream_t stream) {
  const float* hs_pad = (const float*)d_in[0];   // (B,T,EPROJS)
  const int*   ys     = (const int*)  d_in[1];   // (B,U)
  const float* embed  = (const float*)d_in[2];   // (ODIM,EMBED)
  const float* W_ih0  = (const float*)d_in[3];
  const float* W_hh0  = (const float*)d_in[4];
  const float* b_ih0  = (const float*)d_in[5];
  const float* b_hh0  = (const float*)d_in[6];
  const float* W_ih1  = (const float*)d_in[7];
  const float* W_hh1  = (const float*)d_in[8];
  const float* b_ih1  = (const float*)d_in[9];
  const float* b_hh1  = (const float*)d_in[10];
  const float* W_enc  = (const float*)d_in[11];
  const float* b_enc  = (const float*)d_in[12];
  const float* W_dec  = (const float*)d_in[13];
  const float* W_out  = (const float*)d_in[14];
  const float* b_out  = (const float*)d_in[15];
  float* out = (float*)d_out;
  (void)in_sizes; (void)n_in; (void)out_size; (void)ws_size;

  char* wp = (char*)d_ws;
  auto alloc = [&](size_t bytes) -> char* {
    char* p = wp;
    wp += (bytes + 255) & ~(size_t)255;
    return p;
  };
  int*      bar    = (int*)     alloc(256);
  float*    bias0c = (float*)   alloc((size_t)GATES * 4);
  float*    bias1c = (float*)   alloc((size_t)GATES * 4);
  float*    X0     = (float*)   alloc((size_t)B_ * U_ * GATES * 4);
  float*    enc_p  = (float*)   alloc((size_t)B_ * T_ * JOINT_ * 4);
  float*    dec_p  = (float*)   alloc((size_t)B_ * U_ * JOINT_ * 4);
  _Float16* Hbuf   = (_Float16*)alloc((size_t)16 * DUNITS * 2);
  _Float16* eys_h  = (_Float16*)alloc((size_t)B_ * U_ * EMBED_ * 2);
  _Float16* hs_h   = (_Float16*)alloc((size_t)B_ * T_ * EPROJS * 2);
  _Float16* hdec_h = (_Float16*)alloc((size_t)B_ * U_ * DUNITS * 2);
  _Float16* Wih0_h = (_Float16*)alloc((size_t)GATES * EMBED_ * 2);
  _Float16* Whh0_h = (_Float16*)alloc((size_t)GATES * DUNITS * 2);
  _Float16* Wih1_h = (_Float16*)alloc((size_t)GATES * DUNITS * 2);
  _Float16* Whh1_h = (_Float16*)alloc((size_t)GATES * DUNITS * 2);
  _Float16* Wenc_h = (_Float16*)alloc((size_t)JOINT_ * EPROJS * 2);
  _Float16* Wdec_h = (_Float16*)alloc((size_t)JOINT_ * DUNITS * 2);
  _Float16* Wout_h = (_Float16*)alloc((size_t)ODIM_ * JOINT_ * 2);

  // init
  zero_i32_kernel<<<1, 64, 0, stream>>>(bar, 64);
  zero_f16_kernel<<<16, 256, 0, stream>>>(Hbuf, 16 * DUNITS);

  // f32 -> f16 conversions
  cvt_f16_kernel<<<512, 256, 0, stream>>>(W_ih0, Wih0_h, GATES * EMBED_);
  cvt_f16_kernel<<<1024, 256, 0, stream>>>(W_hh0, Whh0_h, GATES * DUNITS);
  cvt_f16_kernel<<<1024, 256, 0, stream>>>(W_ih1, Wih1_h, GATES * DUNITS);
  cvt_f16_kernel<<<1024, 256, 0, stream>>>(W_hh1, Whh1_h, GATES * DUNITS);
  cvt_f16_kernel<<<128, 256, 0, stream>>>(W_enc, Wenc_h, JOINT_ * EPROJS);
  cvt_f16_kernel<<<256, 256, 0, stream>>>(W_dec, Wdec_h, JOINT_ * DUNITS);
  cvt_f16_kernel<<<512, 256, 0, stream>>>(W_out, Wout_h, ODIM_ * JOINT_);
  cvt_f16_kernel<<<128, 256, 0, stream>>>(hs_pad, hs_h, B_ * T_ * EPROJS);

  add_f32_kernel<<<16, 256, 0, stream>>>(b_ih0, b_hh0, bias0c, GATES);
  add_f32_kernel<<<16, 256, 0, stream>>>(b_ih1, b_hh1, bias1c, GATES);
  embed_kernel<<<128, 256, 0, stream>>>(ys, embed, eys_h, B_ * U_ * EMBED_);

  // X0 = eys @ W_ih0^T + (b_ih0 + b_hh0) : (256, 4096), K=512 -> 4096 tiles
  gemm_awt_kernel<<<512, 256, 0, stream>>>(eys_h, Wih0_h, bias0c, X0,
                                           B_ * U_, GATES, EMBED_);

  // persistent 2-layer LSTM scan over U=64 steps
  lstm_scan_kernel<<<8, 256, 0, stream>>>(X0, Whh0_h, Wih1_h, Whh1_h, bias1c,
                                          Hbuf, hdec_h, bar);

  // enc_p = hs_pad @ W_enc^T + b_enc : (512,512) K=512 -> 1024 tiles
  gemm_awt_kernel<<<128, 256, 0, stream>>>(hs_h, Wenc_h, b_enc, enc_p,
                                           B_ * T_, JOINT_, EPROJS);
  // dec_p = h_dec @ W_dec^T : (256,512) K=1024 -> 512 tiles
  gemm_awt_kernel<<<64, 256, 0, stream>>>(hdec_h, Wdec_h, nullptr, dec_p,
                                          B_ * U_, JOINT_, DUNITS);

  // big joint GEMM with fused tanh: one WG per (b,t)
  joint_kernel<<<B_ * T_, 256, 0, stream>>>(enc_p, dec_p, Wout_h, b_out, out);
}
